// NoiseConv1d_35433480192230
// MI455X (gfx1250) — matrix operand order repaired and checked by reading, not compiled, not run on hardware
//
#include <hip/hip_runtime.h>

// Problem sizes (fixed by the reference)
#define BB   8
#define CIN  512
#define COUT 512
#define NN   4096

typedef __attribute__((ext_vector_type(16))) _Float16 v16h;
typedef __attribute__((ext_vector_type(8)))  _Float16 v8h;
typedef __attribute__((ext_vector_type(8)))  float    v8f;

// ---------------------------------------------------------------------------
// Kernel 0: weight prep.  W (f32, row-major [COUT][CIN]) -> fp16 hi/lo planes
// (hi = fp16(W), lo = fp16(W - hi)) + per-row sums for the offset term.
// ---------------------------------------------------------------------------
__global__ void nc_wprep(const float* __restrict__ w,
                         _Float16* __restrict__ wh, _Float16* __restrict__ wl,
                         float* __restrict__ rowsum) {
  int o = blockIdx.x;            // one block per output row
  int t = threadIdx.x;           // 256 threads
  float s = 0.f;
  for (int c = t; c < CIN; c += 256) {
    float v = w[(size_t)o * CIN + c];
    _Float16 h = (_Float16)v;
    wh[(size_t)o * CIN + c] = h;
    wl[(size_t)o * CIN + c] = (_Float16)(v - (float)h);
    s += v;
  }
  __shared__ float red[256];
  red[t] = s; __syncthreads();
  for (int k = 128; k > 0; k >>= 1) {
    if (t < k) red[t] += red[t + k];
    __syncthreads();
  }
  if (t == 0) rowsum[o] = red[0];
}

// ---------------------------------------------------------------------------
// Kernel 1: per-(b,n) channel min/max + quantize + transpose.
// x: [B][CIN][NN] f32 (n contiguous).  Produces xq: [B][NN][CIN] fp16
// (c contiguous -> GEMM B-fragments are contiguous 32B loads), plus
// scale[b][n] and cmin[b][n].
// Tile: 32 columns (n) x all 512 channels, transposed through LDS.
// ---------------------------------------------------------------------------
__global__ void nc_quant(const float* __restrict__ x,
                         _Float16* __restrict__ xq,
                         float* __restrict__ scale, float* __restrict__ cmin) {
  const int b  = blockIdx.y;
  const int n0 = blockIdx.x * 32;
  const int t  = threadIdx.x;                // 256 threads
  __shared__ float smin[8][32], smax[8][32];
  __shared__ float sinv[32], smn[32];
  __shared__ _Float16 ldsT[32][520];         // row stride 520 halves = 1040B = 65*16B

  const float* xb = x + ((size_t)b * CIN) * NN + n0;

  // Phase A: min/max over channels (8 partial reducers x 32 columns)
  int nl = t & 31, cp = t >> 5;
  float mn = 3.4e38f, mx = -3.4e38f;
  for (int c = cp * 64; c < cp * 64 + 64; ++c) {
    float v = xb[(size_t)c * NN + nl];       // coalesced across nl
    mn = fminf(mn, v); mx = fmaxf(mx, v);
  }
  smin[cp][nl] = mn; smax[cp][nl] = mx;
  __syncthreads();
  if (t < 32) {
    float a = smin[0][t], z = smax[0][t];
    for (int j = 1; j < 8; ++j) { a = fminf(a, smin[j][t]); z = fmaxf(z, smax[j][t]); }
    float rng = z - a;
    smn[t]  = a;
    sinv[t] = (rng > 0.f) ? (63.f / rng) : 0.f;
    scale[(size_t)b * NN + n0 + t] = rng * (1.f / 63.f);
    cmin [(size_t)b * NN + n0 + t] = a;
  }
  __syncthreads();

  // Phase B: quantize (re-read is L2-hot), store transposed into LDS
  for (int i = 0; i < 64; ++i) {
    int f = i * 256 + t;
    int n = f & 31, c = f >> 5;              // coalesced reads across n
    float v = xb[(size_t)c * NN + n];
    float q = rintf((v - smn[n]) * sinv[n]); // integer in [0,63], exact in fp16
    ldsT[n][c] = (_Float16)q;
  }
  __syncthreads();

  // Phase C: coalesced 16B stores of the transposed tile
  _Float16* dst = xq + ((size_t)b * NN + n0) * CIN;
  for (int i = 0; i < 8; ++i) {
    int ch = i * 256 + t;                    // 2048 chunks of 8 halves
    int n = ch >> 6, seg = ch & 63;
    v8h v = *(const v8h*)&ldsT[n][seg * 8];  // row base 1040B -> 16B aligned
    *(v8h*)(dst + ((size_t)n) * CIN + seg * 8) = v;
  }
}

// ---------------------------------------------------------------------------
// Fragment loaders.
// A (16-bit 16x32, row-major W): lane l = m + 16*h.
//   halves 0..15 = K[16h .. 16h+15] (contiguous, matches B)
// B (16-bit 32x16, xq is [n][c] so K contiguous per column): lane l = n + 16*h
//   holds K[16h .. 16h+15] contiguously -> single 32B load.
// ---------------------------------------------------------------------------
__device__ __forceinline__ v16h ldA(const _Float16* p) {
  v8h lo = *(const v8h*)p;
  v8h hi = *(const v8h*)(p + 8);
  return __builtin_shufflevector(lo, hi, 0,1,2,3,4,5,6,7,8,9,10,11,12,13,14,15);
}

// ---------------------------------------------------------------------------
// Kernel 2: GEMM + epilogue.  Block = 128 threads (4 waves, 2x2), wave tile
// 32(M) x 64(N): 2 M-frags x 4 N-frags = 8 f32 accumulators.  K = 512 in 16
// steps of 32; hi+lo W planes -> 16 WMMAs per step.
// ---------------------------------------------------------------------------
__global__ __launch_bounds__(128) void nc_gemm(
    const _Float16* __restrict__ wh, const _Float16* __restrict__ wl,
    const _Float16* __restrict__ xq, const float* __restrict__ scale,
    const float* __restrict__ cmin, const float* __restrict__ rowsum,
    const float* __restrict__ bias, float* __restrict__ out) {
  const int t = threadIdx.x;
  const int lane = t & 31, wid = t >> 5;
  const int wm = wid & 1, wn = wid >> 1;
  const int b  = blockIdx.z;
  const int m0 = blockIdx.y * 64 + wm * 32;
  const int n0 = blockIdx.x * 128 + wn * 64;
  const int am = lane & 15, ah = lane >> 4;

  const _Float16* arowh[2] = { wh + (size_t)(m0 + am) * CIN,
                               wh + (size_t)(m0 + 16 + am) * CIN };
  const _Float16* arowl[2] = { wl + (size_t)(m0 + am) * CIN,
                               wl + (size_t)(m0 + 16 + am) * CIN };
  const _Float16* brow[4];
  #pragma unroll
  for (int nf = 0; nf < 4; ++nf)
    brow[nf] = xq + ((size_t)b * NN + n0 + nf * 16 + am) * CIN;

  v8f acc[2][4] = {};

  #pragma unroll 4
  for (int k0 = 0; k0 < CIN; k0 += 32) {
    v16h bf[4];
    #pragma unroll
    for (int nf = 0; nf < 4; ++nf)
      bf[nf] = *(const v16h*)(brow[nf] + k0 + 16 * ah);   // 32B contiguous
    v16h ahi[2], alo[2];
    #pragma unroll
    for (int mf = 0; mf < 2; ++mf) {
      ahi[mf] = ldA(arowh[mf] + k0 + 16 * ah);
      alo[mf] = ldA(arowl[mf] + k0 + 16 * ah);
    }
    #pragma unroll
    for (int mf = 0; mf < 2; ++mf)
      #pragma unroll
      for (int nf = 0; nf < 4; ++nf) {
        acc[mf][nf] = __builtin_amdgcn_wmma_f32_16x16x32_f16(
            false, alo[mf], false, bf[nf], (short)0, acc[mf][nf], false, false);
        acc[mf][nf] = __builtin_amdgcn_wmma_f32_16x16x32_f16(
            false, ahi[mf], false, bf[nf], (short)0, acc[mf][nf], false, false);
      }
  }

  // Epilogue: D layout (16x16 f32): reg r, lanes 0-15 -> (M=r, N=lane),
  // lanes 16-31 -> (M=r+8, N=lane-16).
  #pragma unroll
  for (int nf = 0; nf < 4; ++nf) {
    int n = n0 + nf * 16 + am;
    float sc = scale[(size_t)b * NN + n];
    float mn = cmin [(size_t)b * NN + n];
    #pragma unroll
    for (int mf = 0; mf < 2; ++mf) {
      #pragma unroll
      for (int r = 0; r < 8; ++r) {
        int m = m0 + mf * 16 + r + 8 * ah;
        float val = acc[mf][nf][r] * sc + mn * rowsum[m] + bias[m];
        out[((size_t)b * COUT + m) * NN + n] = val;
      }
    }
  }
}

// ---------------------------------------------------------------------------
// Launcher
// ---------------------------------------------------------------------------
extern "C" void kernel_launch(void* const* d_in, const int* in_sizes, int n_in,
                              void* d_out, int out_size, void* d_ws, size_t ws_size,
                              hipStream_t stream) {
  const float* x    = (const float*)d_in[0];   // [8][512][4096]
  const float* w    = (const float*)d_in[1];   // [512][512][1]
  const float* bias = (const float*)d_in[2];   // [512]
  float* out = (float*)d_out;

  char* ws = (char*)d_ws;
  size_t off = 0;
  auto carve = [&](size_t bytes) -> char* {
    char* p = ws + off;
    off = (off + bytes + 255) & ~(size_t)255;
    return p;
  };
  _Float16* xq   = (_Float16*)carve((size_t)BB * NN * CIN * 2);  // 32 MB
  _Float16* wh   = (_Float16*)carve((size_t)COUT * CIN * 2);
  _Float16* wl   = (_Float16*)carve((size_t)COUT * CIN * 2);
  float*    rsum = (float*)   carve((size_t)COUT * 4);
  float*    sc   = (float*)   carve((size_t)BB * NN * 4);
  float*    cmn  = (float*)   carve((size_t)BB * NN * 4);
  (void)ws_size; (void)in_sizes; (void)n_in; (void)out_size;

  nc_wprep<<<COUT, 256, 0, stream>>>(w, wh, wl, rsum);
  nc_quant<<<dim3(NN / 32, BB), 256, 0, stream>>>(x, xq, sc, cmn);
  nc_gemm<<<dim3(NN / 128, COUT / 64, BB), 128, 0, stream>>>(
      wh, wl, xq, sc, cmn, rsum, bias, out);
}
